// GraphAutoencoder_19086834663831
// MI455X (gfx1250) — matrix operand order repaired and checked by reading, not compiled, not run on hardware
//
#include <hip/hip_runtime.h>
#include <hip/hip_bf16.h>
#include <math.h>

// ---------------- problem constants ----------------
#define NN   50000
#define EE   800000
#define ESL  850000          // EE + NN self loops
#define FEAT_E 16
#define BN_INV 0.9999950000374996f   // 1/sqrt(1+1e-5)

typedef __bf16 bf16_t;
typedef __attribute__((ext_vector_type(16))) __bf16 v16bf;
typedef __attribute__((ext_vector_type(8)))  __bf16 v8bf;
typedef __attribute__((ext_vector_type(8)))  float  v8f;

// order-preserving float<->uint encoding for atomicMax-based segment max
__device__ __forceinline__ unsigned fenc(float f){
  unsigned u = __float_as_uint(f);
  return (u & 0x80000000u) ? ~u : (u | 0x80000000u);
}
__device__ __forceinline__ float fdec(unsigned u){
  return (u & 0x80000000u) ? __uint_as_float(u & 0x7fffffffu) : __uint_as_float(~u);
}

// ---------------- elementwise utility kernels ----------------
__global__ void k_f32_to_bf16(const float* __restrict__ in, bf16_t* __restrict__ out, size_t n){
  size_t i = (size_t)blockIdx.x * blockDim.x + threadIdx.x;
  if (i < n) out[i] = (bf16_t)in[i];
}
__global__ void k_fill_f32(float* __restrict__ p, float v, size_t n){
  size_t i = (size_t)blockIdx.x * blockDim.x + threadIdx.x;
  if (i < n) p[i] = v;
}
__global__ void k_fill_u32(unsigned* __restrict__ p, unsigned v, size_t n){
  size_t i = (size_t)blockIdx.x * blockDim.x + threadIdx.x;
  if (i < n) p[i] = v;
}

// column means of edge_attr [EE, 16]; one block per column
__global__ void k_colmean(const float* __restrict__ ea, float* __restrict__ mean,
                          int rows, int cols){
  __shared__ float sm[256];
  int c = blockIdx.x;
  float acc = 0.f;
  for (int i = threadIdx.x; i < rows; i += blockDim.x) acc += ea[(size_t)i * cols + c];
  sm[threadIdx.x] = acc; __syncthreads();
  for (int s = blockDim.x >> 1; s > 0; s >>= 1){
    if (threadIdx.x < s) sm[threadIdx.x] += sm[threadIdx.x + s];
    __syncthreads();
  }
  if (threadIdx.x == 0) mean[c] = sm[0] / (float)rows;
}

// M[f,h] = sum_d We[f, h*D+d] * ae[h*D+d];  aleloop[h] = sum_f mean[f]*M[f,h]
__global__ void k_attn_prep(const float* __restrict__ We, const float* __restrict__ ae,
                            const float* __restrict__ mean, float* __restrict__ Mm,
                            float* __restrict__ aleloop, int Hh, int Dd){
  int t = threadIdx.x;
  if (t < FEAT_E * Hh){
    int f = t / Hh, h = t % Hh;
    float acc = 0.f;
    for (int d = 0; d < Dd; ++d) acc += We[(size_t)f * Hh * Dd + h * Dd + d] * ae[h * Dd + d];
    Mm[f * Hh + h] = acc;
  }
  __syncthreads();
  if (t < Hh){
    float acc = 0.f;
    for (int f = 0; f < FEAT_E; ++f) acc += mean[f] * Mm[f * Hh + t];
    aleloop[t] = acc;
  }
}

// ale[e,h] = sum_f ea[e,f]*M[f,h]
__global__ void k_ale(const float* __restrict__ ea, const float* __restrict__ Mm,
                      float* __restrict__ ale, int ecnt, int Hh){
  size_t i = (size_t)blockIdx.x * blockDim.x + threadIdx.x;
  if (i >= (size_t)ecnt * Hh) return;
  int e = (int)(i / Hh), h = (int)(i % Hh);
  float acc = 0.f;
  for (int f = 0; f < FEAT_E; ++f) acc += ea[(size_t)e * FEAT_E + f] * Mm[f * Hh + h];
  ale[i] = acc;
}

// per-node attention dots: als[n,h] = xs[n,h,:].a_s[h,:], ald likewise
__global__ void k_node_dots(const float* __restrict__ xs, const float* __restrict__ as_,
                            const float* __restrict__ ad_, float* __restrict__ als,
                            float* __restrict__ ald, int n, int Hh, int Dd){
  size_t i = (size_t)blockIdx.x * blockDim.x + threadIdx.x;
  if (i >= (size_t)n * Hh) return;
  int node = (int)(i / Hh), h = (int)(i % Hh);
  const float* p = xs + (size_t)node * Hh * Dd + h * Dd;
  float a = 0.f, b = 0.f;
  for (int d = 0; d < Dd; ++d){ a += p[d] * as_[h * Dd + d]; b += p[d] * ad_[h * Dd + d]; }
  als[i] = a; ald[i] = b;
}

// pass 1: logits + segment max over dst
__global__ void k_logits_max(const int* __restrict__ src, const int* __restrict__ dst,
                             const float* __restrict__ als, const float* __restrict__ ald,
                             const float* __restrict__ ale, const float* __restrict__ aleloop,
                             float* __restrict__ logits, unsigned* __restrict__ maxu,
                             int ereal, int etot, int Hh){
  int i = blockIdx.x * blockDim.x + threadIdx.x;
  if (i >= etot) return;
  int s = (i < ereal) ? src[i] : (i - ereal);
  int d = (i < ereal) ? dst[i] : (i - ereal);
  for (int h = 0; h < Hh; ++h){
    float le = (i < ereal) ? ale[(size_t)i * Hh + h] : aleloop[h];
    float lg = als[(size_t)s * Hh + h] + ald[(size_t)d * Hh + h] + le;
    lg = (lg > 0.f) ? lg : 0.2f * lg;           // leaky_relu slope 0.2
    logits[(size_t)i * Hh + h] = lg;
    atomicMax(&maxu[(size_t)d * Hh + h], fenc(lg));
  }
}

// pass 2: p = exp(l - max[dst]); segment sum
__global__ void k_exp_sum(const int* __restrict__ dst, float* __restrict__ logits,
                          const unsigned* __restrict__ maxu, float* __restrict__ sum,
                          int ereal, int etot, int Hh){
  int i = blockIdx.x * blockDim.x + threadIdx.x;
  if (i >= etot) return;
  int d = (i < ereal) ? dst[i] : (i - ereal);
  for (int h = 0; h < Hh; ++h){
    float p = __expf(logits[(size_t)i * Hh + h] - fdec(maxu[(size_t)d * Hh + h]));
    logits[(size_t)i * Hh + h] = p;
    atomicAdd(&sum[(size_t)d * Hh + h], p);
  }
}

// pass 3: alpha = p / sum[dst]
__global__ void k_alpha(const int* __restrict__ dst, float* __restrict__ logits,
                        const float* __restrict__ sum, int ereal, int etot, int Hh){
  int i = blockIdx.x * blockDim.x + threadIdx.x;
  if (i >= etot) return;
  int d = (i < ereal) ? dst[i] : (i - ereal);
  for (int h = 0; h < Hh; ++h)
    logits[(size_t)i * Hh + h] /= sum[(size_t)d * Hh + h];
}

// pass 4: out[dst, c] += alpha[e, c/D] * xs[src, c]
__global__ void k_aggregate(const int* __restrict__ src, const int* __restrict__ dst,
                            const float* __restrict__ alpha, const float* __restrict__ xs,
                            float* __restrict__ out, int ereal, int etot, int Hh, int Dd){
  size_t i = (size_t)blockIdx.x * blockDim.x + threadIdx.x;
  int HD = Hh * Dd;
  size_t tot = (size_t)etot * HD;
  if (i >= tot) return;
  int e = (int)(i / HD), c = (int)(i % HD), h = c / Dd;
  int s = (e < ereal) ? src[e] : (e - ereal);
  int d = (e < ereal) ? dst[e] : (e - ereal);
  atomicAdd(&out[(size_t)d * HD + c], alpha[(size_t)e * Hh + h] * xs[(size_t)s * HD + c]);
}

// h = elu(bn(agg + bias)) -> bf16
__global__ void k_bn_elu(const float* __restrict__ in, const float* __restrict__ bias,
                         const float* __restrict__ g, const float* __restrict__ bb,
                         bf16_t* __restrict__ out, size_t rows, int C){
  size_t i = (size_t)blockIdx.x * blockDim.x + threadIdx.x;
  if (i >= rows * (size_t)C) return;
  int c = (int)(i % C);
  float v = in[i] + bias[c];
  v = g[c] * v * BN_INV + bb[c];
  v = (v > 0.f) ? v : (__expf(v) - 1.f);        // elu
  out[i] = (bf16_t)v;
}

// z = agg + bias (in place on d_out) and bf16 copy
__global__ void k_addbias_cvt(float* __restrict__ z, const float* __restrict__ bias,
                              bf16_t* __restrict__ zb, size_t rows, int C){
  size_t i = (size_t)blockIdx.x * blockDim.x + threadIdx.x;
  if (i >= rows * (size_t)C) return;
  int c = (int)(i % C);
  float v = z[i] + bias[c];
  z[i] = v; zb[i] = (bf16_t)v;
}

// ef = [z[src], z[dst], edge_attr]  (bf16, width 80)
__global__ void k_build_ef(const int* __restrict__ src, const int* __restrict__ dst,
                           const bf16_t* __restrict__ zb, const bf16_t* __restrict__ eab,
                           bf16_t* __restrict__ ef, int ecnt){
  size_t i = (size_t)blockIdx.x * blockDim.x + threadIdx.x;
  if (i >= (size_t)ecnt * 80) return;
  int e = (int)(i / 80), c = (int)(i % 80);
  bf16_t v;
  if (c < 32)       v = zb[(size_t)src[e] * 32 + c];
  else if (c < 64)  v = zb[(size_t)dst[e] * 32 + (c - 32)];
  else              v = eab[(size_t)e * FEAT_E + (c - 64)];
  ef[i] = v;
}

// edge_recon = sigmoid(e2 @ ew3 + eb3)
__global__ void k_edge_out(const bf16_t* __restrict__ e2, const float* __restrict__ ew3,
                           const float* __restrict__ eb3, float* __restrict__ out, int ecnt){
  int e = blockIdx.x * blockDim.x + threadIdx.x;
  if (e >= ecnt) return;
  float acc = eb3[0];
  const bf16_t* p = e2 + (size_t)e * 64;
  for (int j = 0; j < 64; ++j) acc += (float)p[j] * ew3[j];
  out[e] = 1.f / (1.f + __expf(-acc));
}

// ---------------- WMMA GEMM (fully templated) ----------------
// C[M,Nc] = epilogue(A[M,K] @ B[K,Nc] [+ bias]); A,B row-major bf16.
// One 16x16 tile per wave, COLS/16 waves per block, grid = (M/16, Nc/COLS).
// B tile staged in LDS transposed (column-contiguous, zero-padded to KR) -> each wave
// builds its B fragment with two ds_load_b128; A uses two global b128 per k-step.
// MODE 0: identity   MODE 1: relu then BN(g,bb)   MODE 2: relu
// BIAS/OUTF/OUTB compile-time -> straight-line epilogue, no pointer checks.
template<int K, int COLS, int MODE, bool BIAS, bool OUTF, bool OUTB>
__global__ void wmma_gemm_t(const bf16_t* __restrict__ A, const bf16_t* __restrict__ B,
                            const float* __restrict__ bias, const float* __restrict__ g,
                            const float* __restrict__ bb,
                            float* __restrict__ Cf, bf16_t* __restrict__ Cb,
                            int M, int Nc){
  constexpr int KR = (K + 31) & ~31;     // K rounded up to WMMA depth
  constexpr int KP = KR + 8;             // padded column stride (bank-conflict-free)
  __shared__ bf16_t Bs[COLS * KP + 8];

  const int tid  = threadIdx.x;
  const int lane = tid & 31;
  const int wv   = tid >> 5;
  const int hf   = lane >> 4;            // lane half
  const int l16  = lane & 15;
  const int colBase = blockIdx.y * COLS;

  // ---- cooperative B tile load: Bs[c][k] = B[k][colBase+c], zero-pad k in [K,KR) ----
  {
    constexpr int ROW8 = COLS / 8;
    for (int idx = tid; idx < KR * ROW8; idx += blockDim.x){
      int k  = idx / ROW8;
      int c8 = (idx % ROW8) * 8;
      if (k < K){
        v8bf q = *(const v8bf*)(B + (size_t)k * Nc + colBase + c8);
#pragma unroll
        for (int j = 0; j < 8; ++j) Bs[(c8 + j) * KP + k] = q[j];
      } else {
#pragma unroll
        for (int j = 0; j < 8; ++j) Bs[(c8 + j) * KP + k] = (bf16_t)0.0f;
      }
    }
  }
  __syncthreads();

  const int col_local = wv * 16 + l16;
  const int col = colBase + col_local;
  const bf16_t* Ap = A + (size_t)(blockIdx.x * 16 + l16) * K;   // this lane's A row
  const bf16_t* Bcol = Bs + col_local * KP;

  v8f c = {0.f, 0.f, 0.f, 0.f, 0.f, 0.f, 0.f, 0.f};
#pragma unroll
  for (int kb = 0; kb < KR; kb += 32){
    v16bf a, b;
    // A chunk 1: a[0..7], k = kb + hf*8 + j   (guards fold at compile time)
    if (kb + 16 <= K){
      v8bf q = *(const v8bf*)(Ap + kb + hf * 8);
#pragma unroll
      for (int j = 0; j < 8; ++j) a[j] = q[j];
    } else if (kb >= K){
#pragma unroll
      for (int j = 0; j < 8; ++j) a[j] = (bf16_t)0.0f;
    } else {
#pragma unroll
      for (int j = 0; j < 8; ++j){ int k = kb + hf * 8 + j; a[j] = (k < K) ? Ap[k] : (bf16_t)0.0f; }
    }
    // A chunk 2: a[8..15], k = kb + 16 + hf*8 + j
    if (kb + 32 <= K){
      v8bf q = *(const v8bf*)(Ap + kb + 16 + hf * 8);
#pragma unroll
      for (int j = 0; j < 8; ++j) a[8 + j] = q[j];
    } else if (kb + 16 >= K){
#pragma unroll
      for (int j = 0; j < 8; ++j) a[8 + j] = (bf16_t)0.0f;
    } else {
#pragma unroll
      for (int j = 0; j < 8; ++j){ int k = kb + 16 + hf * 8 + j; a[8 + j] = (k < K) ? Ap[k] : (bf16_t)0.0f; }
    }
    // B fragment: b[j] = B[kb + hf*8 + j][col], b[8+j] = B[kb + 16 + hf*8 + j][col] -> two ds_load_b128 (Bs zero-padded to KR)
    {
      const bf16_t* p = Bcol + kb + hf * 8;
      v8bf q0 = *(const v8bf*)(p);
      v8bf q1 = *(const v8bf*)(p + 16);
#pragma unroll
      for (int j = 0; j < 8; ++j){ b[j] = q0[j]; b[8 + j] = q1[j]; }
    }
    c = __builtin_amdgcn_wmma_f32_16x16x32_bf16(false, a, false, b, (short)0, c, false, false);
  }

  float bc = 0.f, gc = 0.f, bbc = 0.f;
  if constexpr (BIAS)      bc  = bias[col];
  if constexpr (MODE == 1){ gc = g[col]; bbc = bb[col]; }
#pragma unroll
  for (int r = 0; r < 8; ++r){
    int rr = blockIdx.x * 16 + hf * 8 + r;   // C/D layout: vgpr r -> row r (+8 for upper half)
    float v = c[r];
    if constexpr (BIAS) v += bc;
    if constexpr (MODE == 1){ v = (v > 0.f) ? v : 0.f; v = gc * v * BN_INV + bbc; }
    if constexpr (MODE == 2){ v = (v > 0.f) ? v : 0.f; }
    size_t idx = (size_t)rr * Nc + col;
    if constexpr (OUTF) Cf[idx] = v;
    if constexpr (OUTB) Cb[idx] = (bf16_t)v;
  }
}

template<int K, int COLS, int MODE, bool BIAS, bool OUTF, bool OUTB>
static void launch_gemm_t(const bf16_t* A, const bf16_t* B, const float* bias,
                          const float* g, const float* bb, float* Cf, bf16_t* Cb,
                          int M, int Nc, hipStream_t s){
  dim3 grid((unsigned)(M / 16), (unsigned)(Nc / COLS));
  wmma_gemm_t<K, COLS, MODE, BIAS, OUTF, OUTB>
      <<<grid, dim3(COLS / 16 * 32), 0, s>>>(A, B, bias, g, bb, Cf, Cb, M, Nc);
}

#define GS(n) dim3((unsigned)((((size_t)(n)) + 255) / 256)), dim3(256)

extern "C" void kernel_launch(void* const* d_in, const int* in_sizes, int n_in,
                              void* d_out, int out_size, void* d_ws, size_t ws_size,
                              hipStream_t stream) {
  (void)in_sizes; (void)n_in; (void)out_size; (void)ws_size;
  // ---- inputs (setup_inputs order) ----
  const float* x    = (const float*)d_in[0];
  const float* ea   = (const float*)d_in[1];
  const int*   ei   = (const int*)d_in[2];
  const int* src = ei; const int* dst = ei + EE;
  const float* w0   = (const float*)d_in[3];
  const float* as0  = (const float*)d_in[4];
  const float* ad0  = (const float*)d_in[5];
  const float* we0  = (const float*)d_in[6];
  const float* ae0  = (const float*)d_in[7];
  const float* b0   = (const float*)d_in[8];
  const float* bn0g = (const float*)d_in[9];
  const float* bn0b = (const float*)d_in[10];
  const float* w1   = (const float*)d_in[11];
  const float* as1  = (const float*)d_in[12];
  const float* ad1  = (const float*)d_in[13];
  const float* we1  = (const float*)d_in[14];
  const float* ae1  = (const float*)d_in[15];
  const float* b1   = (const float*)d_in[16];
  const float* nw1  = (const float*)d_in[17];
  const float* nb1  = (const float*)d_in[18];
  const float* nbng = (const float*)d_in[19];
  const float* nbnb = (const float*)d_in[20];
  const float* nw2  = (const float*)d_in[21];
  const float* nb2  = (const float*)d_in[22];
  const float* nw3  = (const float*)d_in[23];
  const float* nb3  = (const float*)d_in[24];
  const float* ew1  = (const float*)d_in[25];
  const float* eb1  = (const float*)d_in[26];
  const float* ebng = (const float*)d_in[27];
  const float* ebnb = (const float*)d_in[28];
  const float* ew2  = (const float*)d_in[29];
  const float* eb2  = (const float*)d_in[30];
  const float* ew3  = (const float*)d_in[31];
  const float* eb3  = (const float*)d_in[32];
  float* dout = (float*)d_out;
  float* z_out   = dout;                       // [NN,32]
  float* nrec    = dout + (size_t)NN * 32;     // [NN,128]
  float* erec    = dout + (size_t)NN * 32 + (size_t)NN * 128;  // [EE]

  // ---- workspace carve ----
  char* wp = (char*)d_ws; size_t off = 0;
  auto carve = [&](size_t bytes) -> char* {
    char* p = wp + off; off += (bytes + 255) & ~(size_t)255; return p;
  };
  bf16_t* xb   = (bf16_t*)carve((size_t)NN * 128 * 2);
  bf16_t* w0b  = (bf16_t*)carve(128 * 256 * 2);
  bf16_t* w1b  = (bf16_t*)carve(256 * 32 * 2);
  bf16_t* nw1b = (bf16_t*)carve(32 * 128 * 2);
  bf16_t* nw2b = (bf16_t*)carve(128 * 64 * 2);
  bf16_t* nw3b = (bf16_t*)carve(64 * 128 * 2);
  bf16_t* ew1b = (bf16_t*)carve(80 * 128 * 2);
  bf16_t* ew2b = (bf16_t*)carve(128 * 64 * 2);
  bf16_t* eab  = (bf16_t*)carve((size_t)EE * FEAT_E * 2);
  float* xs0   = (float*)carve((size_t)NN * 256 * 4);
  float* als0  = (float*)carve((size_t)NN * 4 * 4);
  float* ald0  = (float*)carve((size_t)NN * 4 * 4);
  float* Mm0   = (float*)carve(16 * 4 * 4);
  float* alel0 = (float*)carve(4 * 4);
  float* eamean= (float*)carve(16 * 4);
  float* Mm1   = (float*)carve(16 * 4);
  float* alel1 = (float*)carve(4);
  float* ale0  = (float*)carve((size_t)EE * 4 * 4);
  float* logit0= (float*)carve((size_t)ESL * 4 * 4);
  unsigned* maxu0 = (unsigned*)carve((size_t)NN * 4 * 4);
  float* sum0  = (float*)carve((size_t)NN * 4 * 4);
  float* out0  = (float*)carve((size_t)NN * 256 * 4);
  bf16_t* hb   = (bf16_t*)carve((size_t)NN * 256 * 2);
  float* xs1   = (float*)carve((size_t)NN * 32 * 4);
  float* als1  = (float*)carve((size_t)NN * 4);
  float* ald1  = (float*)carve((size_t)NN * 4);
  float* ale1  = (float*)carve((size_t)EE * 4);
  float* logit1= (float*)carve((size_t)ESL * 4);
  unsigned* maxu1 = (unsigned*)carve((size_t)NN * 4);
  float* sum1  = (float*)carve((size_t)NN * 4);
  bf16_t* zb   = (bf16_t*)carve((size_t)NN * 32 * 2);
  bf16_t* t1b  = (bf16_t*)carve((size_t)NN * 128 * 2);
  bf16_t* t2b  = (bf16_t*)carve((size_t)NN * 64 * 2);
  bf16_t* efb  = (bf16_t*)carve((size_t)EE * 80 * 2);
  bf16_t* e1b  = (bf16_t*)carve((size_t)EE * 128 * 2);
  bf16_t* e2b  = (bf16_t*)carve((size_t)EE * 64 * 2);

  // ---- bf16 conversions ----
  k_f32_to_bf16<<<GS((size_t)NN*128), 0, stream>>>(x, xb, (size_t)NN*128);
  k_f32_to_bf16<<<GS(128*256), 0, stream>>>(w0, w0b, 128*256);
  k_f32_to_bf16<<<GS(256*32), 0, stream>>>(w1, w1b, 256*32);
  k_f32_to_bf16<<<GS(32*128), 0, stream>>>(nw1, nw1b, 32*128);
  k_f32_to_bf16<<<GS(128*64), 0, stream>>>(nw2, nw2b, 128*64);
  k_f32_to_bf16<<<GS(64*128), 0, stream>>>(nw3, nw3b, 64*128);
  k_f32_to_bf16<<<GS(80*128), 0, stream>>>(ew1, ew1b, 80*128);
  k_f32_to_bf16<<<GS(128*64), 0, stream>>>(ew2, ew2b, 128*64);
  k_f32_to_bf16<<<GS((size_t)EE*FEAT_E), 0, stream>>>(ea, eab, (size_t)EE*FEAT_E);

  // ---- attention precompute ----
  k_colmean<<<dim3(16), dim3(256), 0, stream>>>(ea, eamean, EE, FEAT_E);
  k_attn_prep<<<dim3(1), dim3(64), 0, stream>>>(we0, ae0, eamean, Mm0, alel0, 4, 64);
  k_attn_prep<<<dim3(1), dim3(64), 0, stream>>>(we1, ae1, eamean, Mm1, alel1, 1, 32);

  // ---- GAT layer 0 ----
  launch_gemm_t<128,128,0,false,true,false>(xb, w0b, nullptr, nullptr, nullptr, xs0, nullptr, NN, 256, stream);
  k_node_dots<<<GS((size_t)NN*4), 0, stream>>>(xs0, as0, ad0, als0, ald0, NN, 4, 64);
  k_ale<<<GS((size_t)EE*4), 0, stream>>>(ea, Mm0, ale0, EE, 4);
  k_fill_u32<<<GS((size_t)NN*4), 0, stream>>>(maxu0, 0u, (size_t)NN*4);
  k_fill_f32<<<GS((size_t)NN*4), 0, stream>>>(sum0, 0.f, (size_t)NN*4);
  k_fill_f32<<<GS((size_t)NN*256), 0, stream>>>(out0, 0.f, (size_t)NN*256);
  k_logits_max<<<GS(ESL), 0, stream>>>(src, dst, als0, ald0, ale0, alel0, logit0, maxu0, EE, ESL, 4);
  k_exp_sum<<<GS(ESL), 0, stream>>>(dst, logit0, maxu0, sum0, EE, ESL, 4);
  k_alpha<<<GS(ESL), 0, stream>>>(dst, logit0, sum0, EE, ESL, 4);
  k_aggregate<<<GS((size_t)ESL*256), 0, stream>>>(src, dst, logit0, xs0, out0, EE, ESL, 4, 64);
  k_bn_elu<<<GS((size_t)NN*256), 0, stream>>>(out0, b0, bn0g, bn0b, hb, NN, 256);

  // ---- GAT layer 1 (H=1, D=32) ----
  launch_gemm_t<256,32,0,false,true,false>(hb, w1b, nullptr, nullptr, nullptr, xs1, nullptr, NN, 32, stream);
  k_node_dots<<<GS(NN), 0, stream>>>(xs1, as1, ad1, als1, ald1, NN, 1, 32);
  k_ale<<<GS(EE), 0, stream>>>(ea, Mm1, ale1, EE, 1);
  k_fill_u32<<<GS(NN), 0, stream>>>(maxu1, 0u, NN);
  k_fill_f32<<<GS(NN), 0, stream>>>(sum1, 0.f, NN);
  k_fill_f32<<<GS((size_t)NN*32), 0, stream>>>(z_out, 0.f, (size_t)NN*32);
  k_logits_max<<<GS(ESL), 0, stream>>>(src, dst, als1, ald1, ale1, alel1, logit1, maxu1, EE, ESL, 1);
  k_exp_sum<<<GS(ESL), 0, stream>>>(dst, logit1, maxu1, sum1, EE, ESL, 1);
  k_alpha<<<GS(ESL), 0, stream>>>(dst, logit1, sum1, EE, ESL, 1);
  k_aggregate<<<GS((size_t)ESL*32), 0, stream>>>(src, dst, logit1, xs1, z_out, EE, ESL, 1, 32);
  k_addbias_cvt<<<GS((size_t)NN*32), 0, stream>>>(z_out, b1, zb, NN, 32);   // z output final

  // ---- node reconstruction MLP ----
  launch_gemm_t<32,128,1,true,false,true>(zb,  nw1b, nb1, nbng, nbnb, nullptr, t1b, NN, 128, stream);  // relu+bn
  launch_gemm_t<128,64,2,true,false,true>(t1b, nw2b, nb2, nullptr, nullptr, nullptr, t2b, NN, 64, stream); // relu
  launch_gemm_t<64,128,0,true,true,false>(t2b, nw3b, nb3, nullptr, nullptr, nrec, nullptr, NN, 128, stream);

  // ---- edge reconstruction MLP ----
  k_build_ef<<<GS((size_t)EE*80), 0, stream>>>(src, dst, zb, eab, efb, EE);
  launch_gemm_t<80,128,1,true,false,true>(efb, ew1b, eb1, ebng, ebnb, nullptr, e1b, EE, 128, stream);  // K=80 zero-padded to 96
  launch_gemm_t<128,64,2,true,false,true>(e1b, ew2b, eb2, nullptr, nullptr, nullptr, e2b, EE, 64, stream);
  k_edge_out<<<GS(EE), 0, stream>>>(e2b, ew3, eb3, erec, EE);
}
